// Net_50826642980911
// MI455X (gfx1250) — compile-verified, matrix operand order checked by reading, not run on hardware
//
#include <hip/hip_runtime.h>
#include <hip/hip_bf16.h>
#include <cstdint>
#include <cstddef>

// ---------------------------------------------------------------------------
// CDNA5 / gfx1250 WMMA types (wave32). 16x16x32 f16 -> f32:
//   A frag: 16 halves/lane (two contiguous 8-half K chunks),
//   B frag: 16 halves/lane (same chunks along K of a transposed weight row),
//   C/D   : 8 floats/lane, element i -> (M = i + 8*hi, N = lane&15).
// ---------------------------------------------------------------------------
typedef _Float16 v16h __attribute__((ext_vector_type(16)));
typedef _Float16 v8h  __attribute__((ext_vector_type(8)));
typedef float    v8f  __attribute__((ext_vector_type(8)));
typedef int      v4i  __attribute__((ext_vector_type(4)));

#ifndef BN_EPS_F
#define BN_EPS_F 1e-5f
#endif

// ---- CDNA5 async global->LDS staging (ASYNCcnt path), with safe fallback ----
#if defined(__HIP_DEVICE_COMPILE__) && __has_builtin(__builtin_amdgcn_global_load_async_to_lds_b128)
#define G2L_ASYNC 1
typedef __attribute__((address_space(1))) v4i* gv4i_p;   // prints as __device__ int4*
typedef __attribute__((address_space(3))) v4i* lv4i_p;   // prints as __shared__ int4*
#endif

__device__ __forceinline__ void stage_b128(const _Float16* gp, _Float16* lp) {
#if defined(G2L_ASYNC)
  __builtin_amdgcn_global_load_async_to_lds_b128(
      (gv4i_p)(const void*)gp, (lv4i_p)(void*)lp, 0, 0);
#else
  *(v8h*)lp = *(const v8h*)gp;        // global_load_b128 + ds_store_b128
#endif
}

__device__ __forceinline__ void wait_async() {
#if defined(G2L_ASYNC)
#if __has_builtin(__builtin_amdgcn_s_wait_asynccnt)
  __builtin_amdgcn_s_wait_asynccnt(0);
#else
  asm volatile("s_wait_asynccnt 0x0" ::: "memory");
#endif
#endif
}

// ===========================================================================
// Kernels
// ===========================================================================

// f32 -> f16 elementwise (activations).
__global__ void k_cvt_f16(const float* __restrict__ in, _Float16* __restrict__ out,
                          size_t n)
{
  const size_t t = (size_t)blockIdx.x * blockDim.x + threadIdx.x;
  if (t < n) out[t] = (_Float16)in[t];
}

// W[K,N] f32 -> Wt[N,K] f16 (transpose so B fragments are K-contiguous).
__global__ void k_cvt_f16_t(const float* __restrict__ W, _Float16* __restrict__ Wt,
                            int K, int N)
{
  const int t = blockIdx.x * blockDim.x + threadIdx.x;
  if (t >= K * N) return;
  const int n = t / K, k = t % K;
  Wt[t] = (_Float16)W[(size_t)k * N + n];
}

// Fused GEMM: C[M,N] = act(A[M,K] @ W[K,N] + bias).
// A16: f16 row-major [M,K]; Wt16: f16 [N,K] (W transposed).
// Requirements (guaranteed by the network): K % 32 == 0, M % 16 == 0.
// Block = 4 waves sharing one group of NT column tiles. The B panel
// (16*NT cols x K halves, <= 64KB) is staged once per block into LDS by wave 0
// via GLOBAL_LOAD_ASYNC_TO_LDS_B128 (+ s_wait_asynccnt + barrier); the k-loop
// then reads B fragments with ds_load_b128 and A fragments with
// unconditional global_load_b128, NT v_wmma per step, zero divergence.
template <int NT>
__global__ void k_gemm_wmma(const _Float16* __restrict__ A16,
                            const _Float16* __restrict__ Wt16,
                            const float* __restrict__ bias, float* __restrict__ C,
                            int M, int N, int K, int relu)
{
  extern __shared__ _Float16 Bs[];            // [16*NT][K]
  const int lane = threadIdx.x;               // 0..31
  const int wv   = threadIdx.y;               // 0..3
  const int tn   = blockIdx.x;                // group of NT col tiles
  const int row0 = (blockIdx.y * 4 + wv) * 16;
  const int hi   = lane >> 4;                 // half select
  const int r    = lane & 15;
  const int col0 = tn * (16 * NT);

  // ---- cooperative B panel stage (wave 0), async copy to LDS ----
  if (wv == 0) {
    const int kc8 = K >> 3;                   // 16B chunks per column
    const int chunks = 16 * NT * kc8;         // multiple of 32 for all shapes
    for (int ch = lane; ch < chunks; ch += 32) {
      const int col = ch / kc8;
      const int kc  = (ch % kc8) << 3;
      const int gc  = (col0 + col < N) ? (col0 + col) : (N - 1);  // clamp ragged N
      stage_b128(Wt16 + (size_t)gc * K + kc, Bs + (size_t)col * K + kc);
    }
    wait_async();
  }
  __syncthreads();

  if (row0 < M) {                             // wave-uniform (M % 16 == 0)
    const _Float16* __restrict__ Ap = A16 + (size_t)(row0 + r) * K;

    int bcol[NT];
    const _Float16* Bl[NT];
#pragma unroll
    for (int j = 0; j < NT; ++j) {
      bcol[j] = col0 + 16 * j + r;
      Bl[j]   = Bs + (size_t)(16 * j + r) * K;
    }

    const v8f zero = {};
    v8f acc[NT];
#pragma unroll
    for (int j = 0; j < NT; ++j) acc[j] = zero;

    for (int k0 = 0; k0 < K; k0 += 32) {
      if (k0 + 32 < K) {
        // CDNA5 global_prefetch_b8 of next A slab (uniform guard)
        __builtin_prefetch(Ap + k0 + 32, 0, 0);
      }
      const int o0 = k0 + 8 * hi;
      const int o1 = k0 + 16 + 8 * hi;
      const v8h a0 = *(const v8h*)(Ap + o0);
      const v8h a1 = *(const v8h*)(Ap + o1);
      const v16h a = __builtin_shufflevector(a0, a1, 0, 1, 2, 3, 4, 5, 6, 7,
                                             8, 9, 10, 11, 12, 13, 14, 15);
#pragma unroll
      for (int j = 0; j < NT; ++j) {
        const v8h b0 = *(const v8h*)(Bl[j] + o0);   // ds_load_b128
        const v8h b1 = *(const v8h*)(Bl[j] + o1);
        const v16h b = __builtin_shufflevector(b0, b1, 0, 1, 2, 3, 4, 5, 6, 7,
                                               8, 9, 10, 11, 12, 13, 14, 15);
        acc[j] = __builtin_amdgcn_wmma_f32_16x16x32_f16(false, a, false, b,
                                                        (short)0, acc[j], false, false);
      }
    }

#pragma unroll
    for (int j = 0; j < NT; ++j) {
      if (bcol[j] < N) {
        const float bv = bias ? bias[bcol[j]] : 0.0f;
#pragma unroll
        for (int i = 0; i < 8; ++i) {
          const int rr = row0 + i + 8 * hi;   // always < M (M % 16 == 0)
          float vv = acc[j][i] + bv;
          if (relu) vv = fmaxf(vv, 0.0f);
          C[(size_t)rr * N + bcol[j]] = vv;
        }
      }
    }
  }
}

// Plain linear for tiny K (K=3, K=6) where WMMA has no payoff.
__global__ void k_linear_small(const float* __restrict__ A, const float* __restrict__ W,
                               const float* __restrict__ bias, float* __restrict__ C,
                               int M, int N, int K, int relu)
{
  const int t = blockIdx.x * blockDim.x + threadIdx.x;
  if (t >= M * N) return;
  const int m = t / N, n = t % N;
  float s = bias ? bias[n] : 0.0f;
  for (int k = 0; k < K; ++k) s += A[(size_t)m * K + k] * W[(size_t)k * N + n];
  if (relu) s = fmaxf(s, 0.0f);
  C[t] = s;
}

// Per-column mean + biased variance over M rows. grid = N, block = 256.
__global__ void k_colstats(const float* __restrict__ H, int M, int N,
                           float* __restrict__ mean, float* __restrict__ var)
{
  const int n = blockIdx.x;
  float s = 0.0f, s2 = 0.0f;
  for (int m = threadIdx.x; m < M; m += blockDim.x) {
    const float v = H[(size_t)m * N + n];
    s += v; s2 += v * v;
  }
  __shared__ float sh[256], sh2[256];
  sh[threadIdx.x] = s; sh2[threadIdx.x] = s2;
  __syncthreads();
  for (int o = blockDim.x >> 1; o > 0; o >>= 1) {
    if ((int)threadIdx.x < o) {
      sh[threadIdx.x]  += sh[threadIdx.x + o];
      sh2[threadIdx.x] += sh2[threadIdx.x + o];
    }
    __syncthreads();
  }
  if (threadIdx.x == 0) {
    const float mu = sh[0] / (float)M;
    mean[n] = mu;
    var[n]  = sh2[0] / (float)M - mu * mu;
  }
}

// BatchNorm (training-mode batch stats) + ReLU.
__global__ void k_bn_relu(const float* __restrict__ H, const float* __restrict__ mean,
                          const float* __restrict__ var, const float* __restrict__ g,
                          const float* __restrict__ be, float* __restrict__ O,
                          int M, int N)
{
  const int t = blockIdx.x * blockDim.x + threadIdx.x;
  if (t >= M * N) return;
  const int n = t % N;
  const float v = (H[t] - mean[n]) * __frsqrt_rn(var[n] + BN_EPS_F) * g[n] + be[n];
  O[t] = fmaxf(v, 0.0f);
}

// k-NN by brute force + register insertion sort. One thread per dst point.
template <int KB>
__global__ void k_knn(const float* __restrict__ src, int Ns,
                      const float* __restrict__ dst, int Nd,
                      int exclude_self, int append_self, int stride,
                      int* __restrict__ idx)
{
  const int i = blockIdx.x * blockDim.x + threadIdx.x;
  if (i >= Nd) return;
  const float dx = dst[i * 3], dy = dst[i * 3 + 1], dz = dst[i * 3 + 2];
  float bd[KB]; int bi[KB];
#pragma unroll
  for (int t = 0; t < KB; ++t) { bd[t] = 3.4e38f; bi[t] = 0; }
  for (int j = 0; j < Ns; ++j) {
    if (exclude_self && j == i) continue;
    const float ax = src[j * 3] - dx, ay = src[j * 3 + 1] - dy, az = src[j * 3 + 2] - dz;
    const float d = ax * ax + ay * ay + az * az;
    if (d < bd[KB - 1]) {
      int t = KB - 1;
      while (t > 0 && bd[t - 1] > d) { bd[t] = bd[t - 1]; bi[t] = bi[t - 1]; --t; }
      bd[t] = d; bi[t] = j;
    }
  }
  for (int t = 0; t < KB; ++t) idx[(size_t)i * stride + t] = bi[t];
  if (append_self) idx[(size_t)i * stride + KB] = i;
}

// Farthest point sampling, single block, start at index 0. mind = N floats ws.
__global__ void k_fps(const float* __restrict__ pos, int N, int n_out,
                      int* __restrict__ out_idx, float* __restrict__ mind)
{
  const int T = blockDim.x;
  __shared__ float smax[256];
  __shared__ int   simax[256];
  const float px0 = pos[0], py0 = pos[1], pz0 = pos[2];
  for (int i = threadIdx.x; i < N; i += T) {
    const float a = pos[i * 3] - px0, b = pos[i * 3 + 1] - py0, c = pos[i * 3 + 2] - pz0;
    mind[i] = a * a + b * b + c * c;
  }
  if (threadIdx.x == 0) out_idx[0] = 0;
  __syncthreads();
  for (int t = 1; t < n_out; ++t) {
    float lm = -1.0f; int li = 0;
    for (int i = threadIdx.x; i < N; i += T) {
      const float v = mind[i];
      if (v > lm) { lm = v; li = i; }
    }
    smax[threadIdx.x] = lm; simax[threadIdx.x] = li;
    __syncthreads();
    for (int o = T >> 1; o > 0; o >>= 1) {
      if ((int)threadIdx.x < o && smax[threadIdx.x + o] > smax[threadIdx.x]) {
        smax[threadIdx.x] = smax[threadIdx.x + o];
        simax[threadIdx.x] = simax[threadIdx.x + o];
      }
      __syncthreads();
    }
    const int nxt = simax[0];
    if (threadIdx.x == 0) out_idx[t] = nxt;
    const float qx = pos[nxt * 3], qy = pos[nxt * 3 + 1], qz = pos[nxt * 3 + 2];
    for (int i = threadIdx.x; i < N; i += T) {
      const float a = pos[i * 3] - qx, b = pos[i * 3 + 1] - qy, c = pos[i * 3 + 2] - qz;
      const float d = a * a + b * b + c * c;
      if (d < mind[i]) mind[i] = d;
    }
    __syncthreads();
  }
}

__global__ void k_gather_rows(const float* __restrict__ in, const int* __restrict__ idx,
                              float* __restrict__ out, int n, int cols)
{
  const int t = blockIdx.x * blockDim.x + threadIdx.x;
  if (t >= n * cols) return;
  const int i = t / cols, c = t % cols;
  out[t] = in[(size_t)idx[i] * cols + c];
}

// rel[i,j,:] = pos[i] - pos[nbr[i,j]]
__global__ void k_relpos(const float* __restrict__ pos, const int* __restrict__ nbr,
                         int N, int Kn, float* __restrict__ rel)
{
  const int t = blockIdx.x * blockDim.x + threadIdx.x;
  if (t >= N * Kn) return;
  const int i = t / Kn;
  const int nj = nbr[t];
#pragma unroll
  for (int d = 0; d < 3; ++d)
    rel[(size_t)t * 3 + d] = pos[(size_t)i * 3 + d] - pos[(size_t)nj * 3 + d];
}

// gin[i,j,c] = q[i,c] - k[nbr[i,j],c] + delta[i,j,c]
__global__ void k_gamma_in(const float* __restrict__ q, const float* __restrict__ kk,
                           const float* __restrict__ delta, const int* __restrict__ nbr,
                           int N, int Kn, int C, float* __restrict__ out)
{
  const int t = blockIdx.x * blockDim.x + threadIdx.x;
  if (t >= N * Kn * C) return;
  const int i   = t / (Kn * C);
  const int rem = t % (Kn * C);
  const int j   = rem / C;
  const int c   = rem % C;
  out[t] = q[(size_t)i * C + c] - kk[(size_t)nbr[(size_t)i * Kn + j] * C + c] + delta[t];
}

// Fused per-(i,c) softmax over neighbors + weighted combine.
__global__ void k_attn_out(const float* __restrict__ gamma, const float* __restrict__ v,
                           const float* __restrict__ delta, const int* __restrict__ nbr,
                           int N, int Kn, int C, float* __restrict__ out)
{
  const int t = blockIdx.x * blockDim.x + threadIdx.x;
  if (t >= N * C) return;
  const int i = t / C, c = t % C;
  const float* g = gamma + ((size_t)i * Kn) * C + c;
  float mx = -3.4e38f;
  for (int j = 0; j < Kn; ++j) mx = fmaxf(mx, g[(size_t)j * C]);
  float s = 0.0f;
  for (int j = 0; j < Kn; ++j) s += __expf(g[(size_t)j * C] - mx);
  const float inv = 1.0f / s;
  float o = 0.0f;
  for (int j = 0; j < Kn; ++j) {
    const float w  = __expf(g[(size_t)j * C] - mx) * inv;
    const int   nj = nbr[(size_t)i * Kn + j];
    o += w * (v[(size_t)nj * C + c] + delta[((size_t)i * Kn + j) * C + c]);
  }
  out[t] = o;
}

// out[i,c] = max_j h[nbr[i,j],c]   (transition_down pooling)
__global__ void k_max_gather(const float* __restrict__ h, const int* __restrict__ nbr,
                             int Nd, int Kn, int C, float* __restrict__ out)
{
  const int t = blockIdx.x * blockDim.x + threadIdx.x;
  if (t >= Nd * C) return;
  const int i = t / C, c = t % C;
  float m = -3.4e38f;
  for (int j = 0; j < Kn; ++j)
    m = fmaxf(m, h[(size_t)nbr[(size_t)i * Kn + j] * C + c]);
  out[t] = m;
}

// out[i,c] = b[i,c] + IDW-interp of xs at pos[i] from 3 sub neighbors.
__global__ void k_interp_add(const float* __restrict__ xs, const float* __restrict__ pos,
                             const float* __restrict__ pos_sub, const int* __restrict__ idx,
                             const float* __restrict__ b, int N, int C,
                             float* __restrict__ out)
{
  const int t = blockIdx.x * blockDim.x + threadIdx.x;
  if (t >= N * C) return;
  const int i = t / C, c = t % C;
  const float px = pos[i * 3], py = pos[i * 3 + 1], pz = pos[i * 3 + 2];
  float wsum = 0.0f, acc = 0.0f;
#pragma unroll
  for (int j = 0; j < 3; ++j) {
    const int sj = idx[i * 3 + j];
    const float ax = px - pos_sub[sj * 3], ay = py - pos_sub[sj * 3 + 1], az = pz - pos_sub[sj * 3 + 2];
    const float d2 = ax * ax + ay * ay + az * az;
    const float w  = 1.0f / fmaxf(d2, 1e-16f);
    wsum += w;
    acc  += w * xs[(size_t)sj * C + c];
  }
  out[t] = b[t] + acc / wsum;
}

// Row-wise log_softmax (C small: 13).
__global__ void k_logsoftmax(const float* __restrict__ in, float* __restrict__ out,
                             int N, int C)
{
  const int i = blockIdx.x * blockDim.x + threadIdx.x;
  if (i >= N) return;
  const float* x = in + (size_t)i * C;
  float mx = -3.4e38f;
  for (int c = 0; c < C; ++c) mx = fmaxf(mx, x[c]);
  float s = 0.0f;
  for (int c = 0; c < C; ++c) s += __expf(x[c] - mx);
  const float l = __logf(s);
  for (int c = 0; c < C; ++c) out[(size_t)i * C + c] = x[c] - mx - l;
}

// ===========================================================================
// Host-side orchestration
// ===========================================================================

namespace {

constexpr int DIMSV[5] = {32, 64, 128, 256, 512};
constexpr int NLV[5]   = {8192, 2048, 512, 128, 32};
constexpr int KNN_K    = 16;
constexpr int KN1      = KNN_K + 1;   // 17 (knn_graph + self loop)

struct Lin { const float* W; const float* b; int in, out; };
struct BN1 { const float* be; const float* g; Lin lin; };
struct TB  { Lin attn0, attn1, dst, lin_in, lin_out, src, v, pos0, pos1; int c; };
struct TUP { BN1 mlp, mlp_sub; };

// Params arrive flattened in jax pytree order: dict keys sorted alphabetically,
// lists in order. Cursor consumes pointers sequentially.
struct Cur {
  void* const* d; int i;
  const float* nx() { return (const float*)d[i++]; }
};

Lin mk_lin(Cur& p, int ci, int co, bool bias) {
  Lin l; l.W = p.nx(); l.b = bias ? p.nx() : nullptr; l.in = ci; l.out = co; return l;
}
BN1 mk_bn(Cur& p, int ci, int co) {               // keys: be, g, lin{W,b}
  BN1 b; b.be = p.nx(); b.g = p.nx(); b.lin = mk_lin(p, ci, co, true); return b;
}
TB mk_tb(Cur& p, int c) {  // keys: attn_nn, lin_dst, lin_in, lin_out, lin_src, lin_v, pos_nn
  TB t; t.c = c;
  t.attn0   = mk_lin(p, c, 64, true);
  t.attn1   = mk_lin(p, 64, c, true);
  t.dst     = mk_lin(p, c, c, false);
  t.lin_in  = mk_lin(p, c, c, true);
  t.lin_out = mk_lin(p, c, c, true);
  t.src     = mk_lin(p, c, c, false);
  t.v       = mk_lin(p, c, c, false);
  t.pos0    = mk_lin(p, 3, 64, true);
  t.pos1    = mk_lin(p, 64, c, true);
  return t;
}

// Bump allocator over d_ws; pass by value for scoped temporaries. Reuse across
// stream-ordered kernels is safe (one stream serializes producers/consumers).
struct Bump {
  char*  base;
  size_t off;
  float*    f(size_t n) { size_t s = (n * 4 + 255) & ~(size_t)255; float* r = (float*)(base + off); off += s; return r; }
  int*      ip(size_t n){ size_t s = (n * 4 + 255) & ~(size_t)255; int*   r = (int*)(base + off);   off += s; return r; }
  _Float16* h(size_t n) { size_t s = (n * 2 + 255) & ~(size_t)255; _Float16* r = (_Float16*)(base + off); off += s; return r; }
};

inline unsigned cdiv(size_t a, size_t b) { return (unsigned)((a + b - 1) / b); }

// Dense layer: pre-convert A -> f16 and W -> transposed f16, run WMMA GEMM.
void dense(hipStream_t s, Bump ws, const float* A, const Lin& L, float* C, int M, int relu) {
  if (L.in >= 16) {                              // K % 32 == 0 for all such layers
    const size_t na = (size_t)M * L.in;
    const size_t nw = (size_t)L.out * L.in;
    _Float16* A16  = ws.h(na);
    _Float16* Wt16 = ws.h(nw);
    k_cvt_f16  <<<cdiv(na, 256), 256, 0, s>>>(A, A16, na);
    k_cvt_f16_t<<<cdiv(nw, 256), 256, 0, s>>>(L.W, Wt16, L.in, L.out);
    if (L.out >= 64) {
      dim3 g(cdiv(L.out, 64), cdiv(M, 64)), b(32, 4);
      k_gemm_wmma<4><<<g, b, (size_t)64 * L.in * 2, s>>>(A16, Wt16, L.b, C, M, L.out, L.in, relu);
    } else if (L.out >= 32) {
      dim3 g(cdiv(L.out, 32), cdiv(M, 64)), b(32, 4);
      k_gemm_wmma<2><<<g, b, (size_t)32 * L.in * 2, s>>>(A16, Wt16, L.b, C, M, L.out, L.in, relu);
    } else {
      dim3 g(cdiv(L.out, 16), cdiv(M, 64)), b(32, 4);
      k_gemm_wmma<1><<<g, b, (size_t)16 * L.in * 2, s>>>(A16, Wt16, L.b, C, M, L.out, L.in, relu);
    }
  } else {
    const size_t tot = (size_t)M * L.out;
    k_linear_small<<<cdiv(tot, 256), 256, 0, s>>>(A, L.W, L.b, C, M, L.out, L.in, relu);
  }
}

void bn_fwd(hipStream_t s, Bump ws, const float* x, const BN1& p, float* out, int M) {
  const int N = p.lin.out;
  float* h  = ws.f((size_t)M * N);
  float* mv = ws.f((size_t)2 * N);
  dense(s, ws, x, p.lin, h, M, 0);
  k_colstats<<<N, 256, 0, s>>>(h, M, N, mv, mv + N);
  k_bn_relu<<<cdiv((size_t)M * N, 256), 256, 0, s>>>(h, mv, mv + N, p.g, p.be, out, M, N);
}

void tblock_fwd(hipStream_t s, Bump ws, const TB& p, const float* x,
                const float* pos, const int* nbr, int N, int Kn, float* out) {
  const int C = p.c;
  const size_t NK = (size_t)N * Kn;
  float* x1 = ws.f((size_t)N * C); dense(s, ws, x,  p.lin_in, x1, N, 1);
  float* q  = ws.f((size_t)N * C); dense(s, ws, x1, p.dst,    q,  N, 0);
  float* kk = ws.f((size_t)N * C); dense(s, ws, x1, p.src,    kk, N, 0);
  float* v  = ws.f((size_t)N * C); dense(s, ws, x1, p.v,      v,  N, 0);

  float* rel = ws.f(NK * 3);
  k_relpos<<<cdiv(NK, 256), 256, 0, s>>>(pos, nbr, N, Kn, rel);
  float* ph    = ws.f(NK * 64); dense(s, ws, rel, p.pos0, ph,    (int)NK, 1);
  float* delta = ws.f(NK * C);  dense(s, ws, ph,  p.pos1, delta, (int)NK, 1);

  float* gin = ws.f(NK * C);
  k_gamma_in<<<cdiv(NK * C, 256), 256, 0, s>>>(q, kk, delta, nbr, N, Kn, C, gin);
  float* ah = ph;                 // reuse: pos hidden dead after delta
  dense(s, ws, gin, p.attn0, ah, (int)NK, 1);
  float* gamma = gin;             // reuse: gin dead after ah
  dense(s, ws, ah, p.attn1, gamma, (int)NK, 1);

  float* at = x1;                 // reuse: x1 dead after q/k/v
  k_attn_out<<<cdiv((size_t)N * C, 256), 256, 0, s>>>(gamma, v, delta, nbr, N, Kn, C, at);
  dense(s, ws, at, p.lin_out, out, N, 1);
}

void tdown_fwd(hipStream_t s, Bump ws, const BN1& p, const float* x, const float* pos,
               int N, float* x_out, float* pos_out) {
  const int n_out = N / 4;
  float* mind = ws.f((size_t)N);
  int*   idx  = ws.ip((size_t)n_out);
  k_fps<<<1, 256, 0, s>>>(pos, N, n_out, idx, mind);
  k_gather_rows<<<cdiv((size_t)n_out * 3, 256), 256, 0, s>>>(pos, idx, pos_out, n_out, 3);
  int* nbr = ws.ip((size_t)n_out * KNN_K);
  k_knn<KNN_K><<<cdiv(n_out, 256), 256, 0, s>>>(pos, N, pos_out, n_out, 0, 0, KNN_K, nbr);
  float* h = ws.f((size_t)N * p.lin.out);
  bn_fwd(s, ws, x, p, h, N);
  k_max_gather<<<cdiv((size_t)n_out * p.lin.out, 256), 256, 0, s>>>(h, nbr, n_out, KNN_K, p.lin.out, x_out);
}

void tup_fwd(hipStream_t s, Bump ws, const TUP& p, const float* x_skip, const float* x_sub,
             const float* pos, const float* pos_sub, int N, int N_sub, float* out) {
  const int cl = p.mlp.lin.out;
  float* a = ws.f((size_t)N_sub * cl); bn_fwd(s, ws, x_sub,  p.mlp_sub, a, N_sub);
  float* b = ws.f((size_t)N * cl);     bn_fwd(s, ws, x_skip, p.mlp,     b, N);
  int* i3 = ws.ip((size_t)N * 3);
  k_knn<3><<<cdiv(N, 256), 256, 0, s>>>(pos_sub, N_sub, pos, N, 0, 0, 3, i3);
  k_interp_add<<<cdiv((size_t)N * cl, 256), 256, 0, s>>>(a, pos, pos_sub, i3, b, N, cl, out);
}

} // namespace

extern "C" void kernel_launch(void* const* d_in, const int* in_sizes, int n_in,
                              void* d_out, int out_size, void* d_ws, size_t ws_size,
                              hipStream_t stream) {
  (void)in_sizes; (void)n_in; (void)out_size; (void)ws_size;
  hipStream_t s = stream;

  // ---- parse params (flatten order: sorted dict keys, lists in order) ----
  Cur p{d_in, 2};
  BN1 mlp_input = mk_bn(p, 6, DIMSV[0]);                           // "mlp_input"
  Lin out0 = mk_lin(p, DIMSV[0], 64, true);                        // "mlp_output"
  Lin out1 = mk_lin(p, 64, 64, true);
  Lin out2 = mk_lin(p, 64, 13, true);
  Lin summit = mk_lin(p, DIMSV[4], DIMSV[4], true);                // "mlp_summit"
  TB t_down[4]; for (int i = 0; i < 4; ++i) t_down[i] = mk_tb(p, DIMSV[i + 1]); // "t_down"
  TB t_in = mk_tb(p, DIMSV[0]);                                    // "t_in"
  TB t_summit = mk_tb(p, DIMSV[4]);                                // "t_summit"
  TB t_up[4]; for (int i = 0; i < 4; ++i) t_up[i] = mk_tb(p, DIMSV[i]);         // "t_up"
  BN1 td[4]; for (int i = 0; i < 4; ++i) td[i] = mk_bn(p, DIMSV[i], DIMSV[i + 1]); // "td"
  TUP tu[4];                                                       // "tu": keys mlp < mlp_sub
  for (int i = 0; i < 4; ++i) {
    tu[i].mlp     = mk_bn(p, DIMSV[i], DIMSV[i]);
    tu[i].mlp_sub = mk_bn(p, DIMSV[i + 1], DIMSV[i]);
  }

  // ---- persistent buffers (skips, positions, graphs, up-path) ----
  Bump ws{(char*)d_ws, 0};
  float* xs[5]; int* nbrs[5]; float* poss[5]; float* xu[4];
  poss[0] = (float*)d_in[1];                     // read-only input positions
  for (int l = 0; l < 5; ++l) {
    xs[l]   = ws.f((size_t)NLV[l] * DIMSV[l]);
    nbrs[l] = ws.ip((size_t)NLV[l] * KN1);
    if (l > 0) poss[l] = ws.f((size_t)NLV[l] * 3);
  }
  for (int l = 0; l < 4; ++l) xu[l] = ws.f((size_t)NLV[l] * DIMSV[l]);

  // ---- input MLP + entry transformer block ----
  {
    Bump w2 = ws;
    float* xa = w2.f((size_t)NLV[0] * DIMSV[0]);
    bn_fwd(s, w2, (const float*)d_in[0], mlp_input, xa, NLV[0]);
    k_knn<KNN_K><<<cdiv(NLV[0], 256), 256, 0, s>>>(poss[0], NLV[0], poss[0], NLV[0], 1, 1, KN1, nbrs[0]);
    tblock_fwd(s, w2, t_in, xa, poss[0], nbrs[0], NLV[0], KN1, xs[0]);
  }

  // ---- encoder: transition_down + tblock, levels 1..4 ----
  float* xc = xs[0];
  for (int l = 0; l < 4; ++l) {
    Bump w2 = ws;
    float* xd = w2.f((size_t)NLV[l + 1] * DIMSV[l + 1]);
    tdown_fwd(s, w2, td[l], xc, poss[l], NLV[l], xd, poss[l + 1]);
    k_knn<KNN_K><<<cdiv(NLV[l + 1], 256), 256, 0, s>>>(poss[l + 1], NLV[l + 1],
                                                       poss[l + 1], NLV[l + 1], 1, 1, KN1, nbrs[l + 1]);
    tblock_fwd(s, w2, t_down[l], xd, poss[l + 1], nbrs[l + 1], NLV[l + 1], KN1, xs[l + 1]);
    xc = xs[l + 1];
  }

  // ---- summit ----
  {
    Bump w2 = ws;
    float* hs = w2.f((size_t)NLV[4] * DIMSV[4]);
    dense(s, w2, xs[4], summit, hs, NLV[4], 1);
    tblock_fwd(s, w2, t_summit, hs, poss[4], nbrs[4], NLV[4], KN1, xs[4]);
    xc = xs[4];
  }

  // ---- decoder: transition_up + tblock, levels 3..0 ----
  for (int i = 0; i < 4; ++i) {
    const int lskip = 3 - i;
    const int lsub  = 4 - i;
    Bump w2 = ws;
    float* xt = w2.f((size_t)NLV[lskip] * DIMSV[lskip]);
    tup_fwd(s, w2, tu[lskip], xs[lskip], xc, poss[lskip], poss[lsub],
            NLV[lskip], NLV[lsub], xt);
    tblock_fwd(s, w2, t_up[lskip], xt, poss[lskip], nbrs[lskip], NLV[lskip], KN1, xu[lskip]);
    xc = xu[lskip];
  }

  // ---- output head + log_softmax ----
  {
    Bump w2 = ws;
    float* h1 = w2.f((size_t)NLV[0] * 64); dense(s, w2, xc, out0, h1, NLV[0], 1);
    float* h2 = w2.f((size_t)NLV[0] * 64); dense(s, w2, h1, out1, h2, NLV[0], 1);
    float* h3 = w2.f((size_t)NLV[0] * 13); dense(s, w2, h2, out2, h3, NLV[0], 0);
    k_logsoftmax<<<cdiv(NLV[0], 256), 256, 0, s>>>(h3, (float*)d_out, NLV[0], 13);
  }
}
